// MultiHeadAttentionLayer_72404558676375
// MI455X (gfx1250) — matrix operand order repaired and checked by reading, not compiled, not run on hardware
//
#include <hip/hip_runtime.h>
#include <stdint.h>
#include <stddef.h>

// B=4, N=256, IN_DIM=128, IN_DIM_E=32, H=8, D=16, H*D=128
typedef __bf16 bf16_t;
typedef __attribute__((ext_vector_type(16))) __bf16 v16bf;
typedef __attribute__((ext_vector_type(8)))  __bf16 v8bf;
typedef __attribute__((ext_vector_type(8)))  float  v8f;

__device__ __forceinline__ bf16_t f2bf(float f) {
  union { float f; unsigned u; } x; x.f = f;
  unsigned u = x.u + 0x7FFFu + ((x.u >> 16) & 1u);   // round-to-nearest-even
  unsigned short hs = (unsigned short)(u >> 16);
  bf16_t r; __builtin_memcpy(&r, &hs, sizeof(r));
  return r;
}

// monotone float->uint mapping so integer atomicMax == float max (handles negatives)
__device__ __forceinline__ unsigned fkey(float f) {
  union { float f; unsigned u; } x; x.f = f;
  return (x.u & 0x80000000u) ? ~x.u : (x.u | 0x80000000u);
}
__device__ __forceinline__ float funkey(unsigned u) {
  union { float f; unsigned u; } x;
  x.u = (u & 0x80000000u) ? (u & 0x7FFFFFFFu) : ~u;
  return x.f;
}

__device__ __forceinline__ v16bf cat16(v8bf lo, v8bf hi) {
  return __builtin_shufflevector(lo, hi, 0,1,2,3,4,5,6,7,8,9,10,11,12,13,14,15);
}

// ---------------------------------------------------------------- prep: f32->bf16 copies + init
__global__ void __launch_bounds__(256) k_prep(
    const float* __restrict__ h,  const float* __restrict__ Wq,
    const float* __restrict__ Wk, const float* __restrict__ Wv,
    const float* __restrict__ We,
    bf16_t* __restrict__ hbf, bf16_t* __restrict__ Wqb, bf16_t* __restrict__ Wkb,
    bf16_t* __restrict__ Wvb, bf16_t* __restrict__ Web, unsigned* __restrict__ gmaxu) {
  int gid = blockIdx.x * 256 + threadIdx.x;
  if (gid < 131072) hbf[gid] = f2bf(h[gid]);           // 1024 x 128
  if (gid < 16384) {                                   // 128 x 128 each
    Wqb[gid] = f2bf(Wq[gid]);
    Wkb[gid] = f2bf(Wk[gid]);
    Wvb[gid] = f2bf(Wv[gid]);
  }
  if (gid < 4096)  Web[gid] = f2bf(We[gid]);           // 32 x 128
  if (gid < 32)    gmaxu[gid] = 0u;                    // maps to -max
}

// ---------------------------------------------------------------- QKV projections via WMMA
// grid (64 row-tiles, 3 matrices), 256 threads (8 waves = 8 column tiles)
__global__ void __launch_bounds__(256) k_qkv(
    const bf16_t* __restrict__ hbf,
    const bf16_t* __restrict__ Wqb, const bf16_t* __restrict__ Wkb,
    const bf16_t* __restrict__ Wvb,
    float* __restrict__ Qh, float* __restrict__ Khs, bf16_t* __restrict__ Vbf) {
  __shared__ __align__(16) bf16_t sA[16][128];    // 16 rows of h (bf16)
  __shared__ __align__(16) bf16_t sBT[128][128];  // W transposed: [col][k]
  const int rt  = blockIdx.x;        // row tile 0..63 (global rows rt*16..)
  const int w   = blockIdx.y;        // 0=Q 1=K 2=V
  const int tid = threadIdx.x;
  const bf16_t* Wb = (w == 0) ? Wqb : (w == 1) ? Wkb : Wvb;

  #pragma unroll
  for (int m = 0; m < 8; ++m) {
    int idx = tid + m * 256;                       // 0..2047
    sA[idx >> 7][idx & 127] = hbf[rt * 16 * 128 + idx];
  }
  #pragma unroll
  for (int m = 0; m < 64; ++m) {
    int idx = tid + m * 256;                       // 0..16383, k=idx>>7, col=idx&127
    sBT[idx & 127][idx >> 7] = Wb[idx];
  }
  __syncthreads();

  const int lane  = tid & 31;
  const int ct    = tid >> 5;                      // column tile (head) 0..7
  const int rl    = lane & 15;
  const int base0 = (lane < 16) ? 0 : 8;           // A K-offset / D row-offset
  const int kb    = (lane < 16) ? 0 : 16;          // unused; B loads also use base0 as K-offset
  v8f acc = {};
  #pragma unroll
  for (int ks = 0; ks < 4; ++ks) {
    v8bf alo = *reinterpret_cast<const v8bf*>(&sA[rl][ks * 32 + base0]);
    v8bf ahi = *reinterpret_cast<const v8bf*>(&sA[rl][ks * 32 + 16 + base0]);
    v8bf blo = *reinterpret_cast<const v8bf*>(&sBT[ct * 16 + rl][ks * 32 + base0]);
    v8bf bhi = *reinterpret_cast<const v8bf*>(&sBT[ct * 16 + rl][ks * 32 + 16 + base0]);
    acc = __builtin_amdgcn_wmma_f32_16x16x32_bf16(false, cat16(alo, ahi),
                                                  false, cat16(blo, bhi),
                                                  (short)0, acc, false, false);
  }
  #pragma unroll
  for (int r = 0; r < 8; ++r) {
    int row = rt * 16 + r + base0;                 // global row (b*256 + n)
    int col = ct * 16 + rl;                        // h*16 + d
    if (w == 0)      Qh[row * 128 + col]  = acc[r];
    else if (w == 1) Khs[row * 128 + col] = acc[r] * 0.25f;   // D^-0.5
    else {
      int b = row >> 8, j = row & 255;
      Vbf[(((b * 8 + ct) * 256) + j) * 16 + rl] = f2bf(acc[r]); // [b][h][j][d]
    }
  }
}

// ---------------------------------------------------------------- fused E-GEMM + triple contraction
// one block per (b,i); Esl = e[b,i,:,:] @ We via WMMA, then scores[h,j] = sum_d Esl*Kh*Qh
__global__ void __launch_bounds__(256) k_scores(
    const float* __restrict__ e, const float* __restrict__ Qh,
    const float* __restrict__ Khs, const bf16_t* __restrict__ Web,
    float* __restrict__ scores, unsigned* __restrict__ gmaxu) {
  __shared__ __align__(16) bf16_t sE[256][32];     // e slice (bf16) [j][c]
  __shared__ __align__(16) bf16_t sWeT[128][32];   // We transposed [col][c]
  __shared__ float    sQ[128];
  __shared__ unsigned smax[8];
  const int bi = blockIdx.x;                       // b*256 + i
  const int b = bi >> 8, i = bi & 255;
  const int tid = threadIdx.x;

  const float* eb = e + (size_t)bi * (256 * 32);
  #pragma unroll
  for (int m = 0; m < 32; ++m) {
    int idx = tid + m * 256;                       // 0..8191
    sE[idx >> 5][idx & 31] = f2bf(eb[idx]);
  }
  #pragma unroll
  for (int m = 0; m < 16; ++m) {
    int idx = tid + m * 256;                       // 0..4095, k=idx>>7, col=idx&127
    sWeT[idx & 127][idx >> 7] = Web[idx];
  }
  if (tid < 128) sQ[tid] = Qh[(size_t)bi * 128 + tid];
  if (tid < 8)   smax[tid] = 0u;
  __syncthreads();

  const int wave  = tid >> 5, lane = tid & 31;
  const int rl    = lane & 15;
  const int base0 = (lane < 16) ? 0 : 8;
  const int kb    = (lane < 16) ? 0 : 16;

  for (int t = wave; t < 128; t += 8) {            // 16 j-tiles x 8 heads
    const int jt = t >> 3, hh = t & 7;
    v8bf alo = *reinterpret_cast<const v8bf*>(&sE[jt * 16 + rl][base0]);
    v8bf ahi = *reinterpret_cast<const v8bf*>(&sE[jt * 16 + rl][16 + base0]);
    v8bf blo = *reinterpret_cast<const v8bf*>(&sWeT[hh * 16 + rl][base0]);
    v8bf bhi = *reinterpret_cast<const v8bf*>(&sWeT[hh * 16 + rl][16 + base0]);
    v8f cc = {};
    cc = __builtin_amdgcn_wmma_f32_16x16x32_bf16(false, cat16(alo, ahi),
                                                 false, cat16(blo, bhi),
                                                 (short)0, cc, false, false);
    const float q = sQ[hh * 16 + rl];              // Qh[b,i,hh,d], d = rl
    #pragma unroll
    for (int r = 0; r < 8; ++r) {
      int j = jt * 16 + r + base0;
      float kv = Khs[((size_t)(b * 256 + j)) * 128 + hh * 16 + rl];
      float p = cc[r] * kv * q;                    // per-d contribution
      p += __shfl_xor(p, 1, 32);                   // reduce over the 16 d-lanes
      p += __shfl_xor(p, 2, 32);
      p += __shfl_xor(p, 4, 32);
      p += __shfl_xor(p, 8, 32);
      if (rl == 0) {
        scores[(((size_t)(b * 8 + hh) * 256 + i) * 256) + j] = p;
        atomicMax(&smax[hh], fkey(p));
      }
    }
  }
  __syncthreads();
  if (tid < 8) atomicMax(&gmaxu[b * 8 + tid], smax[tid]);
}

// ---------------------------------------------------------------- exp/gate + P@V via WMMA
// one block per (b,h); 8 waves, each wave owns 2 i-tiles (private LDS panel)
__global__ void __launch_bounds__(256) k_out(
    const float* __restrict__ scores, const bf16_t* __restrict__ Vbf,
    const float* __restrict__ kRW, const float* __restrict__ mask,
    const unsigned* __restrict__ gmaxu, float* __restrict__ out) {
  __shared__ __align__(16) bf16_t sVT[16][256];        // V transposed [d][j]
  __shared__ __align__(16) bf16_t pBuf[8][16][128];    // per-wave P half-panel
  __shared__ float dnm[8][16];
  const int bh = blockIdx.x;                            // b*8 + h
  const int b = bh >> 3, hh = bh & 7;
  const int tid = threadIdx.x;

  #pragma unroll
  for (int m = 0; m < 16; ++m) {
    int idx = tid + m * 256;                           // 0..4095, j=idx>>4, d=idx&15
    sVT[idx & 15][idx >> 4] = Vbf[(size_t)bh * 4096 + idx];
  }
  const float mx = funkey(gmaxu[bh]);
  __syncthreads();

  const int wave  = tid >> 5, lane = tid & 31;
  const int rl    = lane & 15;
  const int base0 = (lane < 16) ? 0 : 8;
  const int kb    = (lane < 16) ? 0 : 16;

  for (int itb = 0; itb < 2; ++itb) {
    const int it = wave * 2 + itb;                     // i-tile 0..15
    if (lane < 16) dnm[wave][lane] = 0.f;
    v8f acc = {};
    for (int jh = 0; jh < 2; ++jh) {                   // j in halves of 128
      for (int rr = 0; rr < 16; ++rr) {
        const int i = it * 16 + rr;
        float partial = 0.f;
        #pragma unroll
        for (int mm = 0; mm < 4; ++mm) {
          int jj = lane + mm * 32;                     // 0..127
          int j  = jh * 128 + jj;
          float s = scores[((size_t)bh * 256 + i) * 256 + j];
          float p = __expf(s - mx) * mask[b * 256 + i] * mask[b * 256 + j]
                    * kRW[((size_t)b * 256 + i) * 256 + j];
          pBuf[wave][rr][jj] = f2bf(p);
          partial += p;
        }
        partial += __shfl_xor(partial, 1, 32);
        partial += __shfl_xor(partial, 2, 32);
        partial += __shfl_xor(partial, 4, 32);
        partial += __shfl_xor(partial, 8, 32);
        partial += __shfl_xor(partial, 16, 32);
        if (lane == 0) dnm[wave][rr] += partial;       // deterministic, single lane
      }
      #pragma unroll
      for (int ks = 0; ks < 4; ++ks) {
        v8bf alo = *reinterpret_cast<const v8bf*>(&pBuf[wave][rl][ks * 32 + base0]);
        v8bf ahi = *reinterpret_cast<const v8bf*>(&pBuf[wave][rl][ks * 32 + 16 + base0]);
        v8bf blo = *reinterpret_cast<const v8bf*>(&sVT[rl][jh * 128 + ks * 32 + base0]);
        v8bf bhi = *reinterpret_cast<const v8bf*>(&sVT[rl][jh * 128 + ks * 32 + 16 + base0]);
        acc = __builtin_amdgcn_wmma_f32_16x16x32_bf16(false, cat16(alo, ahi),
                                                      false, cat16(blo, bhi),
                                                      (short)0, acc, false, false);
      }
    }
    #pragma unroll
    for (int r = 0; r < 8; ++r) {
      int il = r + base0;
      int i  = it * 16 + il;
      float den = fmaxf(dnm[wave][il], 1e-6f);
      out[((size_t)b * 256 + i) * 128 + hh * 16 + rl] = acc[r] / den;
    }
  }
}

// ---------------------------------------------------------------- host launcher
extern "C" void kernel_launch(void* const* d_in, const int* in_sizes, int n_in,
                              void* d_out, int out_size, void* d_ws, size_t ws_size,
                              hipStream_t stream) {
  const float* h    = (const float*)d_in[0];
  const float* e    = (const float*)d_in[1];
  const float* kRW  = (const float*)d_in[2];
  const float* mask = (const float*)d_in[3];
  const float* Wq   = (const float*)d_in[4];
  const float* Wk   = (const float*)d_in[5];
  const float* We   = (const float*)d_in[6];
  const float* Wv   = (const float*)d_in[7];
  float* out = (float*)d_out;

  char* ws = (char*)d_ws;
  size_t o = 0;
  auto take = [&](size_t bytes) -> char* {
    char* p = ws + o;
    o += (bytes + 255) & ~(size_t)255;
    return p;
  };
  bf16_t*   hbf    = (bf16_t*)take(131072 * 2);      // h in bf16
  bf16_t*   Wqb    = (bf16_t*)take(16384 * 2);
  bf16_t*   Wkb    = (bf16_t*)take(16384 * 2);
  bf16_t*   Wvb    = (bf16_t*)take(16384 * 2);
  bf16_t*   Web    = (bf16_t*)take(4096 * 2);
  float*    Qh     = (float*)take(131072 * 4);       // [b*256+i][h*16+d]
  float*    Khs    = (float*)take(131072 * 4);       // pre-scaled by D^-0.5
  bf16_t*   Vbf    = (bf16_t*)take(131072 * 2);      // [b][h][j][d]
  unsigned* gmaxu  = (unsigned*)take(32 * 4);        // per (b,h) score max (keyed)
  float*    scoresW= (float*)take((size_t)2097152 * 4); // [b][h][i][j]

  k_prep<<<512, 256, 0, stream>>>(h, Wq, Wk, Wv, We, hbf, Wqb, Wkb, Wvb, Web, gmaxu);
  k_qkv<<<dim3(64, 3), 256, 0, stream>>>(hbf, Wqb, Wkb, Wvb, Qh, Khs, Vbf);
  k_scores<<<1024, 256, 0, stream>>>(e, Qh, Khs, Web, scoresW, gmaxu);
  k_out<<<32, 256, 0, stream>>>(scoresW, Vbf, kRW, mask, gmaxu, out);
}
